// ImportanceAttentionLayer_55714315764417
// MI455X (gfx1250) — matrix operand order repaired and checked by reading, not compiled, not run on hardware
//
#include <hip/hip_runtime.h>
#include <hip/hip_bf16.h>
#include <cstdint>
#include <cstddef>

typedef __bf16 bf16;
typedef __attribute__((ext_vector_type(16))) __bf16 v16bf;
typedef __attribute__((ext_vector_type(8)))  __bf16 v8bf;
typedef __attribute__((ext_vector_type(8)))  float  v8f;
typedef __attribute__((ext_vector_type(4)))  float  v4f;

#define S_LEN   2048
#define HDIM    1024
#define NHEADS  16
#define HEAD_D  64
#define MLP_DIM 32

__device__ __forceinline__ v8f wmma_bf16(v16bf a, v16bf b, v8f c) {
  // v_wmma_f32_16x16x32_bf16: D = A(16x32) * B(32x16) + C
  return __builtin_amdgcn_wmma_f32_16x16x32_bf16(false, a, false, b, (short)0, c, false, false);
}

// Build a 16-element bf16 fragment from two contiguous 16B chunks (ds_load_b128 / global_load_b128)
__device__ __forceinline__ v16bf ld_frag(const bf16* lo, const bf16* hi) {
  v8bf a = *(const v8bf*)lo;
  v8bf b = *(const v8bf*)hi;
  v16bf r;
#pragma unroll
  for (int i = 0; i < 8; ++i) { r[i] = a[i]; r[i + 8] = b[i]; }
  return r;
}

// Async global->LDS copy of 16B per lane (GLOBAL_LOAD_ASYNC_TO_LDS_B128, ASYNCcnt).
// IOFFSET is added to BOTH the LDS and global addresses (ISA 07_vmem §async), so a
// shared immediate offset covers symmetric copies.
#define ASYNC_CP16(lds_addr_u32, gaddr_u64, OFF_STR)                                 \
  asm volatile("global_load_async_to_lds_b128 %0, %1, off" OFF_STR                   \
               :: "v"(lds_addr_u32), "v"(gaddr_u64) : "memory")

__device__ __forceinline__ unsigned lds_off_u32(const void* p) {
  // LDS aperture: hardware uses addr[31:0] as the LDS byte address
  return (unsigned)(uintptr_t)p;
}

__device__ __forceinline__ float gelu_exact(float x) {
  return 0.5f * x * (1.0f + erff(x * 0.70710678118654752f));
}

// ---------------------------------------------------------------------------
// Kernel 0: importance-score MLPs -> reach[b,h,s], attract[b,h,s]
// ---------------------------------------------------------------------------
__global__ __launch_bounds__(256) void mlp_kernel(
    const float* __restrict__ imp,
    const float* __restrict__ rw1, const float* __restrict__ rb1,
    const float* __restrict__ rw2, const float* __restrict__ rb2,
    const float* __restrict__ aw1, const float* __restrict__ ab1,
    const float* __restrict__ aw2, const float* __restrict__ ab2,
    float* __restrict__ reach, float* __restrict__ attract)
{
  int idx = blockIdx.x * 256 + threadIdx.x;       // b*S + s, < 4096
  if (idx >= 2 * S_LEN) return;
  int b = idx >> 11, s = idx & (S_LEN - 1);
  float x = imp[idx];
  float r[NHEADS], a[NHEADS];
#pragma unroll
  for (int h = 0; h < NHEADS; ++h) { r[h] = rb2[h]; a[h] = ab2[h]; }
#pragma unroll 4
  for (int j = 0; j < MLP_DIM; ++j) {
    float hr = gelu_exact(x * rw1[j] + rb1[j]);
    float ha = gelu_exact(x * aw1[j] + ab1[j]);
#pragma unroll
    for (int h = 0; h < NHEADS; ++h) {
      r[h] += hr * rw2[j * NHEADS + h];
      a[h] += ha * aw2[j * NHEADS + h];
    }
  }
#pragma unroll
  for (int h = 0; h < NHEADS; ++h) {
    reach  [((size_t)(b * NHEADS + h)) * S_LEN + s] = r[h];
    attract[((size_t)(b * NHEADS + h)) * S_LEN + s] = a[h];
  }
}

// ---------------------------------------------------------------------------
// Kernel 1: fused QKV projection GEMM (bf16 WMMA, fp32 accumulate)
// grid = (N/128, M/128, 3); block = 256 (8 waves; wave -> 32x64 C subtile)
// Q,K stored [b,h,s,d] bf16 ; V stored transposed [b,h,d,s] bf16
// ---------------------------------------------------------------------------
__global__ __launch_bounds__(256) void qkv_gemm_kernel(
    const float* __restrict__ X,
    const float* __restrict__ Wq, const float* __restrict__ Bq,
    const float* __restrict__ Wk, const float* __restrict__ Bk,
    const float* __restrict__ Wv, const float* __restrict__ Bv,
    bf16* __restrict__ qb, bf16* __restrict__ kb, bf16* __restrict__ vb)
{
  const int z = blockIdx.z;
  const float* W  = (z == 0) ? Wq : (z == 1) ? Wk : Wv;
  const float* Bi = (z == 0) ? Bq : (z == 1) ? Bk : Bv;
  const int n0 = blockIdx.x * 128;
  const int m0 = blockIdx.y * 128;
  __shared__ alignas(16) bf16 As[128][32];   // M-major, K contiguous
  __shared__ alignas(16) bf16 Bt[128][32];   // N-major (transposed W tile), K contiguous
  const int tid = threadIdx.x;
  const int lane = tid & 31, w = tid >> 5;
  const int r = lane & 15, hh = lane >> 4;
  const int wm = (w & 3) * 32, wn = (w >> 2) * 64;
  const v8f vz = {0.f, 0.f, 0.f, 0.f, 0.f, 0.f, 0.f, 0.f};
  v8f acc[2][4];
#pragma unroll
  for (int mt = 0; mt < 2; ++mt)
#pragma unroll
    for (int nt = 0; nt < 4; ++nt) acc[mt][nt] = vz;

  const int arow = tid >> 1, ac = (tid & 1) * 16;
  const int bkk  = tid >> 3, bnn = (tid & 7) * 16;

  for (int k0 = 0; k0 < HDIM; k0 += 32) {
    // stage A tile (fp32 -> bf16)
    {
      const float* src = X + (size_t)(m0 + arow) * HDIM + k0 + ac;
      __builtin_prefetch((const void*)(src + 32), 0, 1);   // next k-tile
#pragma unroll
      for (int i = 0; i < 16; i += 4) {
        v4f v = *(const v4f*)(src + i);
        As[arow][ac + i + 0] = (bf16)v[0];
        As[arow][ac + i + 1] = (bf16)v[1];
        As[arow][ac + i + 2] = (bf16)v[2];
        As[arow][ac + i + 3] = (bf16)v[3];
      }
    }
    // stage B tile transposed (W is K-major; LDS gets N-major)
    {
      const float* src = W + (size_t)(k0 + bkk) * HDIM + n0 + bnn;
      __builtin_prefetch((const void*)(src + (size_t)32 * HDIM), 0, 1);
#pragma unroll
      for (int i = 0; i < 16; i += 4) {
        v4f v = *(const v4f*)(src + i);
        Bt[bnn + i + 0][bkk] = (bf16)v[0];
        Bt[bnn + i + 1][bkk] = (bf16)v[1];
        Bt[bnn + i + 2][bkk] = (bf16)v[2];
        Bt[bnn + i + 3][bkk] = (bf16)v[3];
      }
    }
    __syncthreads();
    v16bf afr[2];
#pragma unroll
    for (int mt = 0; mt < 2; ++mt)
      afr[mt] = ld_frag(&As[wm + mt * 16 + r][hh * 8],
                        &As[wm + mt * 16 + r][16 + hh * 8]);
#pragma unroll
    for (int nt = 0; nt < 4; ++nt) {
      v16bf bfr = ld_frag(&Bt[wn + nt * 16 + r][hh * 8],
                          &Bt[wn + nt * 16 + r][16 + hh * 8]);
#pragma unroll
      for (int mt = 0; mt < 2; ++mt)
        acc[mt][nt] = wmma_bf16(afr[mt], bfr, acc[mt][nt]);
    }
    __syncthreads();
  }
  // epilogue: +bias, scatter into attention layouts
#pragma unroll
  for (int nt = 0; nt < 4; ++nt) {
    const int n = n0 + wn + nt * 16 + r;
    const float bias = Bi[n];
    const int h = n >> 6, d = n & (HEAD_D - 1);
#pragma unroll
    for (int mt = 0; mt < 2; ++mt) {
#pragma unroll
      for (int e = 0; e < 8; ++e) {
        const int m = m0 + wm + mt * 16 + hh * 8 + e;
        const int b = m >> 11, s = m & (S_LEN - 1);
        const float val = acc[mt][nt][e] + bias;
        if (z == 2)
          vb[(((size_t)(b * NHEADS + h)) * HEAD_D + d) * S_LEN + s] = (bf16)val;
        else if (z == 0)
          qb[(((size_t)(b * NHEADS + h)) * S_LEN + s) * HEAD_D + d] = (bf16)val;
        else
          kb[(((size_t)(b * NHEADS + h)) * S_LEN + s) * HEAD_D + d] = (bf16)val;
      }
    }
  }
}

// ---------------------------------------------------------------------------
// Kernel 2: flash attention with learned mask bias
// grid = (S/128, B*NH); block = 256 (8 waves; wave owns 16 q rows)
// K/V^T tiles staged with GLOBAL_LOAD_ASYNC_TO_LDS_B128 (ASYNCcnt)
// ---------------------------------------------------------------------------
__global__ __launch_bounds__(256) void flash_attn_kernel(
    const bf16* __restrict__ qb, const bf16* __restrict__ kb,
    const bf16* __restrict__ vb,
    const float* __restrict__ reach, const float* __restrict__ attract,
    const float* __restrict__ log_temp, float* __restrict__ attn_out)
{
  const int bh = blockIdx.y;
  const int h = bh & (NHEADS - 1);
  const int b = bh >> 4;
  const int q0 = blockIdx.x * 128;
  __shared__ alignas(16) bf16 Kt[128][HEAD_D];    // kv-major (B-op layout for Q*K^T)
  __shared__ alignas(16) bf16 Vt[HEAD_D][128];    // d-major  (B-op layout for P*V)
  __shared__ alignas(16) float attr_s[S_LEN];
  __shared__ float reach_s[128];
  __shared__ alignas(16) bf16 Pbuf[8][16][128];   // per-wave P staging

  const int tid = threadIdx.x;
  const int lane = tid & 31, w = tid >> 5;
  const int r = lane & 15, hh = lane >> 4;

  // stage attract row (8KB) asynchronously: 32B per lane
  {
    const unsigned la = lds_off_u32(&attr_s[0]) + (unsigned)tid * 32u;
    const uint64_t ga = (uint64_t)(uintptr_t)(attract + (size_t)bh * S_LEN) + (uint64_t)tid * 32u;
    ASYNC_CP16(la, ga, "");
    ASYNC_CP16(la, ga, " offset:16");
  }
  if (tid < 128) reach_s[tid] = reach[(size_t)bh * S_LEN + q0 + tid];

  const float temp = __expf(log_temp[h]) + 1.0f;
  const float inv_st = 1.0f / ((float)S_LEN * temp);

  // Q fragments live in registers for the whole kernel (16 rows x 64 d per wave)
  const bf16* qrow = qb + (((size_t)bh * S_LEN) + q0 + w * 16 + r) * HEAD_D;
  v16bf qf[2];
  qf[0] = ld_frag(qrow + hh * 8,      qrow + 16 + hh * 8);
  qf[1] = ld_frag(qrow + 32 + hh * 8, qrow + 48 + hh * 8);

  float mrow[8], lrow[8];
#pragma unroll
  for (int e = 0; e < 8; ++e) { mrow[e] = -1e30f; lrow[e] = 0.0f; }
  const v8f vz = {0.f, 0.f, 0.f, 0.f, 0.f, 0.f, 0.f, 0.f};
  v8f Oa[4];
#pragma unroll
  for (int dt = 0; dt < 4; ++dt) Oa[dt] = vz;

  for (int j = 0; j < S_LEN / 128; ++j) {
    const int kv0 = j * 128;
    __syncthreads();                       // previous iteration's readers are done

    // stage K tile (16KB, contiguous) : async, 16B/lane x 4 rounds via IOFFSET
    {
      const unsigned la = lds_off_u32(&Kt[0][0]) + (unsigned)tid * 16u;
      const uint64_t ga = (uint64_t)(uintptr_t)(kb + (((size_t)bh * S_LEN) + kv0) * HEAD_D)
                        + (uint64_t)tid * 16u;
      ASYNC_CP16(la, ga, "");
      ASYNC_CP16(la, ga, " offset:4096");
      ASYNC_CP16(la, ga, " offset:8192");
      ASYNC_CP16(la, ga, " offset:12288");
    }
    // stage V^T tile (16KB): vb already [b,h,d,s]; 64B of one row per lane
    {
      const int d = tid >> 2, seg = (tid & 3) * 32;      // 32 bf16 = 64B
      const unsigned la = lds_off_u32(&Vt[0][0]) + (unsigned)(d * 256 + seg * 2);
      const uint64_t ga = (uint64_t)(uintptr_t)(vb + (((size_t)bh * HEAD_D) + d) * S_LEN + kv0 + seg);
      ASYNC_CP16(la, ga, "");
      ASYNC_CP16(la, ga, " offset:16");
      ASYNC_CP16(la, ga, " offset:32");
      ASYNC_CP16(la, ga, " offset:48");
    }
    asm volatile("s_wait_asynccnt 0" ::: "memory");   // own async copies landed in LDS
    __syncthreads();                                   // all waves' copies landed

    // S = Q * K^T  (16 x 128 per wave, fp32 acc)
    v8f sacc[8];
#pragma unroll
    for (int nt = 0; nt < 8; ++nt) {
      const bf16* krow = &Kt[nt * 16 + r][0];
      v16bf b0 = ld_frag(krow + hh * 8,      krow + 16 + hh * 8);       // d 0..31
      v16bf b1 = ld_frag(krow + 32 + hh * 8, krow + 48 + hh * 8);  // d 32..63
      v8f c = vz;
      c = wmma_bf16(qf[0], b0, c);
      c = wmma_bf16(qf[1], b1, c);
      sacc[nt] = c;
    }

    // scale + learned mask bias + online softmax
#pragma unroll
    for (int e = 0; e < 8; ++e) {
      const int ql = w * 16 + hh * 8 + e;
      const int qg = q0 + ql;
      const float re = reach_s[ql];
      float sv[8];
      float vmax = -1e30f;
#pragma unroll
      for (int nt = 0; nt < 8; ++nt) {
        const int kg = kv0 + nt * 16 + r;
        const float ml = re + attr_s[kg] - fabsf((float)(qg - kg)) * inv_st;
        const float sig = 1.0f / (1.0f + __expf(-ml));
        const float bias = __logf(sig + 1e-8f);
        const float s = sacc[nt][e] * 0.125f + bias;   // 1/sqrt(64)
        sv[nt] = s;
        vmax = fmaxf(vmax, s);
      }
      vmax = fmaxf(vmax, __shfl_xor(vmax, 1));
      vmax = fmaxf(vmax, __shfl_xor(vmax, 2));
      vmax = fmaxf(vmax, __shfl_xor(vmax, 4));
      vmax = fmaxf(vmax, __shfl_xor(vmax, 8));
      const float mnew = fmaxf(mrow[e], vmax);
      const float alpha = __expf(mrow[e] - mnew);
      mrow[e] = mnew;
      float rs = 0.0f;
#pragma unroll
      for (int nt = 0; nt < 8; ++nt) {
        const float p = __expf(sv[nt] - mnew);
        rs += p;
        Pbuf[w][hh * 8 + e][nt * 16 + r] = (bf16)p;
      }
      rs += __shfl_xor(rs, 1);
      rs += __shfl_xor(rs, 2);
      rs += __shfl_xor(rs, 4);
      rs += __shfl_xor(rs, 8);
      lrow[e] = lrow[e] * alpha + rs;
#pragma unroll
      for (int dt = 0; dt < 4; ++dt) Oa[dt][e] *= alpha;
    }
    asm volatile("s_wait_dscnt 0" ::: "memory");   // own P writes visible to own wave

    // O += P * V  (A = P 16x128 via LDS, B = V^T tile)
    v16bf pf[4];
#pragma unroll
    for (int kt = 0; kt < 4; ++kt) {
      const bf16* prow = &Pbuf[w][r][kt * 32];
      pf[kt] = ld_frag(prow + hh * 8, prow + 16 + hh * 8);
    }
#pragma unroll
    for (int dt = 0; dt < 4; ++dt) {
#pragma unroll
      for (int kt = 0; kt < 4; ++kt) {
        const bf16* vrow = &Vt[dt * 16 + r][kt * 32];
        v16bf vf = ld_frag(vrow + hh * 8, vrow + 16 + hh * 8);
        Oa[dt] = wmma_bf16(pf[kt], vf, Oa[dt]);
      }
    }
  }

  // epilogue: normalize, write attn_out back in [b,s,H] layout
#pragma unroll
  for (int dt = 0; dt < 4; ++dt) {
#pragma unroll
    for (int e = 0; e < 8; ++e) {
      const int srow = q0 + w * 16 + hh * 8 + e;
      const float val = Oa[dt][e] / lrow[e];
      attn_out[((size_t)b * S_LEN + srow) * HDIM + h * HEAD_D + dt * 16 + r] = val;
    }
  }
}

// ---------------------------------------------------------------------------
// Kernel 3: output projection GEMM (attn_out @ Wo + bo) -> proj (fp32)
// ---------------------------------------------------------------------------
__global__ __launch_bounds__(256) void out_proj_gemm(
    const float* __restrict__ A, const float* __restrict__ W,
    const float* __restrict__ Bi, float* __restrict__ C)
{
  const int n0 = blockIdx.x * 128;
  const int m0 = blockIdx.y * 128;
  __shared__ alignas(16) bf16 As[128][32];
  __shared__ alignas(16) bf16 Bt[128][32];
  const int tid = threadIdx.x;
  const int lane = tid & 31, w = tid >> 5;
  const int r = lane & 15, hh = lane >> 4;
  const int wm = (w & 3) * 32, wn = (w >> 2) * 64;
  const v8f vz = {0.f, 0.f, 0.f, 0.f, 0.f, 0.f, 0.f, 0.f};
  v8f acc[2][4];
#pragma unroll
  for (int mt = 0; mt < 2; ++mt)
#pragma unroll
    for (int nt = 0; nt < 4; ++nt) acc[mt][nt] = vz;

  const int arow = tid >> 1, ac = (tid & 1) * 16;
  const int bkk  = tid >> 3, bnn = (tid & 7) * 16;

  for (int k0 = 0; k0 < HDIM; k0 += 32) {
    {
      const float* src = A + (size_t)(m0 + arow) * HDIM + k0 + ac;
      __builtin_prefetch((const void*)(src + 32), 0, 1);
#pragma unroll
      for (int i = 0; i < 16; i += 4) {
        v4f v = *(const v4f*)(src + i);
        As[arow][ac + i + 0] = (bf16)v[0];
        As[arow][ac + i + 1] = (bf16)v[1];
        As[arow][ac + i + 2] = (bf16)v[2];
        As[arow][ac + i + 3] = (bf16)v[3];
      }
    }
    {
      const float* src = W + (size_t)(k0 + bkk) * HDIM + n0 + bnn;
      __builtin_prefetch((const void*)(src + (size_t)32 * HDIM), 0, 1);
#pragma unroll
      for (int i = 0; i < 16; i += 4) {
        v4f v = *(const v4f*)(src + i);
        Bt[bnn + i + 0][bkk] = (bf16)v[0];
        Bt[bnn + i + 1][bkk] = (bf16)v[1];
        Bt[bnn + i + 2][bkk] = (bf16)v[2];
        Bt[bnn + i + 3][bkk] = (bf16)v[3];
      }
    }
    __syncthreads();
    v16bf afr[2];
#pragma unroll
    for (int mt = 0; mt < 2; ++mt)
      afr[mt] = ld_frag(&As[wm + mt * 16 + r][hh * 8],
                        &As[wm + mt * 16 + r][16 + hh * 8]);
#pragma unroll
    for (int nt = 0; nt < 4; ++nt) {
      v16bf bfr = ld_frag(&Bt[wn + nt * 16 + r][hh * 8],
                          &Bt[wn + nt * 16 + r][16 + hh * 8]);
#pragma unroll
      for (int mt = 0; mt < 2; ++mt)
        acc[mt][nt] = wmma_bf16(afr[mt], bfr, acc[mt][nt]);
    }
    __syncthreads();
  }
#pragma unroll
  for (int nt = 0; nt < 4; ++nt) {
    const int n = n0 + wn + nt * 16 + r;
    const float bias = Bi[n];
#pragma unroll
    for (int mt = 0; mt < 2; ++mt) {
#pragma unroll
      for (int e = 0; e < 8; ++e) {
        const int m = m0 + wm + mt * 16 + hh * 8 + e;
        C[(size_t)m * HDIM + n] = acc[mt][nt][e] + bias;
      }
    }
  }
}

// ---------------------------------------------------------------------------
// Kernel 4: residual + LayerNorm
// ---------------------------------------------------------------------------
__global__ __launch_bounds__(256) void resid_ln_kernel(
    const float* __restrict__ hs, const float* __restrict__ proj,
    const float* __restrict__ g, const float* __restrict__ be,
    float* __restrict__ out)
{
  const int row = blockIdx.x;
  const int tid = threadIdx.x;
  __shared__ float red[256];
  const float* xr = hs   + (size_t)row * HDIM;
  const float* pr = proj + (size_t)row * HDIM;
  v4f xa = *(const v4f*)(xr + tid * 4);
  v4f pa = *(const v4f*)(pr + tid * 4);
  float x[4];
#pragma unroll
  for (int i = 0; i < 4; ++i) x[i] = xa[i] + pa[i];
  float s = x[0] + x[1] + x[2] + x[3];
  red[tid] = s;
  __syncthreads();
  for (int o = 128; o > 0; o >>= 1) {
    if (tid < o) red[tid] += red[tid + o];
    __syncthreads();
  }
  const float mu = red[0] * (1.0f / HDIM);
  __syncthreads();
  float v = 0.0f;
#pragma unroll
  for (int i = 0; i < 4; ++i) { float d = x[i] - mu; v += d * d; }
  red[tid] = v;
  __syncthreads();
  for (int o = 128; o > 0; o >>= 1) {
    if (tid < o) red[tid] += red[tid + o];
    __syncthreads();
  }
  const float inv = rsqrtf(red[0] * (1.0f / HDIM) + 1e-5f);
  const int c = tid * 4;
#pragma unroll
  for (int i = 0; i < 4; ++i)
    out[(size_t)row * HDIM + c + i] = (x[i] - mu) * inv * g[c + i] + be[c + i];
}

// ---------------------------------------------------------------------------
extern "C" void kernel_launch(void* const* d_in, const int* in_sizes, int n_in,
                              void* d_out, int out_size, void* d_ws, size_t ws_size,
                              hipStream_t stream) {
  (void)in_sizes; (void)n_in; (void)out_size; (void)ws_size;
  const float* hs   = (const float*)d_in[0];
  const float* imp  = (const float*)d_in[1];
  const float* wq   = (const float*)d_in[2];
  const float* bq   = (const float*)d_in[3];
  const float* wk   = (const float*)d_in[4];
  const float* bk   = (const float*)d_in[5];
  const float* wv   = (const float*)d_in[6];
  const float* bv   = (const float*)d_in[7];
  const float* wo   = (const float*)d_in[8];
  const float* bo   = (const float*)d_in[9];
  const float* rw1  = (const float*)d_in[10];
  const float* rb1  = (const float*)d_in[11];
  const float* rw2  = (const float*)d_in[12];
  const float* rb2  = (const float*)d_in[13];
  const float* aw1  = (const float*)d_in[14];
  const float* ab1  = (const float*)d_in[15];
  const float* aw2  = (const float*)d_in[16];
  const float* ab2  = (const float*)d_in[17];
  const float* logt = (const float*)d_in[18];
  const float* lng  = (const float*)d_in[19];
  const float* lnb  = (const float*)d_in[20];

  char* ws = (char*)d_ws;
  // [0, 8Mi)   qb bf16  | reused as proj fp32 after flash (16Mi spans qb+kb)
  // [8Mi,16Mi) kb bf16
  // [16Mi,24Mi) vb bf16 (transposed [b,h,d,s])
  // [24Mi,40Mi) attn_out fp32
  // [40Mi, +256Ki) reach ; [+256Ki, +512Ki) attract
  bf16*  qbuf    = (bf16*)(ws);
  bf16*  kbuf    = (bf16*)(ws + ((size_t)8 << 20));
  bf16*  vbuf    = (bf16*)(ws + ((size_t)16 << 20));
  float* attn_o  = (float*)(ws + ((size_t)24 << 20));
  float* proj    = (float*)(ws);                       // reuse q/k region
  float* reach   = (float*)(ws + ((size_t)40 << 20));
  float* attract = (float*)(ws + ((size_t)40 << 20) + ((size_t)256 << 10));

  mlp_kernel<<<16, 256, 0, stream>>>(imp, rw1, rb1, rw2, rb2, aw1, ab1, aw2, ab2,
                                     reach, attract);
  qkv_gemm_kernel<<<dim3(HDIM / 128, (2 * S_LEN) / 128, 3), 256, 0, stream>>>(
      hs, wq, bq, wk, bk, wv, bv, qbuf, kbuf, vbuf);
  flash_attn_kernel<<<dim3(S_LEN / 128, 2 * NHEADS), 256, 0, stream>>>(
      qbuf, kbuf, vbuf, reach, attract, logt, attn_o);
  out_proj_gemm<<<dim3(HDIM / 128, (2 * S_LEN) / 128), 256, 0, stream>>>(
      attn_o, wo, bo, proj);
  resid_ln_kernel<<<2 * S_LEN, 256, 0, stream>>>(hs, proj, lng, lnb, (float*)d_out);
}
